// MLP_AE_72756745994415
// MI455X (gfx1250) — compile-verified, matrix operand order checked by reading, not run on hardware
//
#include <hip/hip_runtime.h>

#define B_SZ 512
#define D_SZ 512
#define H_SZ 256
#define NOUT 99999

typedef __attribute__((ext_vector_type(16))) _Float16 v16h;
typedef __attribute__((ext_vector_type(8)))  _Float16 v8h;
typedef __attribute__((ext_vector_type(8)))  float    v8f;

// Tensor Data Mover: ROCm 7.2 (clang-22) has a 5-arg builtin, the
// amdgpu-toolchain (clang-23) a 6-arg one (extra int32x8 before cpol).
#if defined(__has_builtin)
#if __has_builtin(__builtin_amdgcn_tensor_load_to_lds)
#define USE_TDM 1
#endif
#endif

#ifdef USE_TDM
typedef unsigned int u32x4 __attribute__((ext_vector_type(4)));
typedef int          i32x4 __attribute__((ext_vector_type(4)));
typedef int          i32x8 __attribute__((ext_vector_type(8)));
#endif

// Load 16 f32 values from a row-major [row, K] matrix into the CDNA5 16-bit
// A/B operand layout: halves[0..7] = p[0..7], halves[8..15] = p[16..23].
// p must already include the per-lane koff (0 or 8) and k0.
__device__ __forceinline__ v16h load_f32_as_op(const float* __restrict__ p) {
    float4 f0 = *(const float4*)(p);
    float4 f1 = *(const float4*)(p + 4);
    float4 f2 = *(const float4*)(p + 16);
    float4 f3 = *(const float4*)(p + 20);
    v16h a;
    a[0]  = (_Float16)f0.x; a[1]  = (_Float16)f0.y; a[2]  = (_Float16)f0.z; a[3]  = (_Float16)f0.w;
    a[4]  = (_Float16)f1.x; a[5]  = (_Float16)f1.y; a[6]  = (_Float16)f1.z; a[7]  = (_Float16)f1.w;
    a[8]  = (_Float16)f2.x; a[9]  = (_Float16)f2.y; a[10] = (_Float16)f2.z; a[11] = (_Float16)f2.w;
    a[12] = (_Float16)f3.x; a[13] = (_Float16)f3.y; a[14] = (_Float16)f3.z; a[15] = (_Float16)f3.w;
    return a;
}

// h = LeakyReLU(x @ w2^T + b2), stored as f16 [512, 256]
__global__ void layer1_wmma(const float* __restrict__ x, const float* __restrict__ w2,
                            const float* __restrict__ b2, _Float16* __restrict__ h16) {
    int tid  = threadIdx.x;
    int wave = blockIdx.x * 8 + (tid >> 5);   // 512 waves total
    int lane = tid & 31;
    int mt = wave >> 4, nt = wave & 15;       // 32 x 16 tiles of 16x16
    int m0 = mt * 16, n0 = nt * 16;
    int lrow = lane & 15;
    int koff = (lane < 16) ? 0 : 8;
    const float* ax = x  + (size_t)(m0 + lrow) * D_SZ + koff;
    const float* bw = w2 + (size_t)(n0 + lrow) * D_SZ + koff;
    v8f c = {};
    for (int k0 = 0; k0 < D_SZ; k0 += 32) {
        v16h a = load_f32_as_op(ax + k0);
        v16h b = load_f32_as_op(bw + k0);
        c = __builtin_amdgcn_wmma_f32_16x16x32_f16(false, a, false, b, (short)0, c, false, false);
    }
    int n = n0 + lrow;
    float bias = b2[n];
    int mbase = m0 + ((lane < 16) ? 0 : 8);
#pragma unroll
    for (int r = 0; r < 8; ++r) {
        float v = c[r] + bias;
        v = (v >= 0.f) ? v : 0.01f * v;       // LeakyReLU(0.01)
        h16[(size_t)(mbase + r) * H_SZ + n] = (_Float16)v;
    }
}

// z = h @ w3^T + b3, f32 [512, 55] (N padded to 64 tiles; cols >= 55 dropped)
__global__ void layer2_wmma(const _Float16* __restrict__ h16, const float* __restrict__ w3,
                            const float* __restrict__ b3, float* __restrict__ z) {
    int tid  = threadIdx.x;
    int wave = blockIdx.x * 8 + (tid >> 5);   // 128 waves total
    int lane = tid & 31;
    int mt = wave >> 2, nt = wave & 3;        // 32 x 4 tiles
    int m0 = mt * 16, n0 = nt * 16;
    int lrow = lane & 15;
    int koff = (lane < 16) ? 0 : 8;
    int col  = n0 + lrow;
    int ccol = (col < 55) ? col : 54;         // clamp: garbage cols never stored
    const _Float16* ah = h16 + (size_t)(m0 + lrow) * H_SZ + koff;
    const float*    bw = w3  + (size_t)ccol * H_SZ + koff;
    v8f c = {};
    for (int k0 = 0; k0 < H_SZ; k0 += 32) {
        v8h lo = *(const v8h*)(ah + k0);
        v8h hi = *(const v8h*)(ah + k0 + 16);
        v16h a;
#pragma unroll
        for (int t = 0; t < 8; ++t) { a[t] = lo[t]; a[8 + t] = hi[t]; }
        v16h b = load_f32_as_op(bw + k0);
        c = __builtin_amdgcn_wmma_f32_16x16x32_f16(false, a, false, b, (short)0, c, false, false);
    }
    if (col < 55) {
        float bias = b3[col];
        int mbase = m0 + ((lane < 16) ? 0 : 8);
#pragma unroll
        for (int r = 0; r < 8; ++r)
            z[(size_t)(mbase + r) * 55 + col] = c[r] + bias;
    }
}

// x3[b, :] = [softmax(z[b,0:5]), softmax per 10-wide group of z[b,5:55]]
__global__ void softmax_kernel(const float* __restrict__ z, float* __restrict__ x3) {
    int b = blockIdx.x * blockDim.x + threadIdx.x;
    if (b >= B_SZ) return;
    const float* zr = z + b * 55;
    float* xr = x3 + b * 55;
    {
        float v[5]; float m = -1e30f;
#pragma unroll
        for (int t = 0; t < 5; ++t) { v[t] = zr[t]; m = fmaxf(m, v[t]); }
        float s = 0.f;
#pragma unroll
        for (int t = 0; t < 5; ++t) { v[t] = __expf(v[t] - m); s += v[t]; }
        float inv = 1.f / s;
#pragma unroll
        for (int t = 0; t < 5; ++t) xr[t] = v[t] * inv;
    }
#pragma unroll
    for (int g = 0; g < 5; ++g) {
        const float* zg = zr + 5 + 10 * g;
        float* xg = xr + 5 + 10 * g;
        float w[10]; float m = -1e30f;
#pragma unroll
        for (int t = 0; t < 10; ++t) { w[t] = zg[t]; m = fmaxf(m, w[t]); }
        float s = 0.f;
#pragma unroll
        for (int t = 0; t < 10; ++t) { w[t] = __expf(w[t] - m); s += w[t]; }
        float inv = 1.f / s;
#pragma unroll
        for (int t = 0; t < 10; ++t) xg[t] = w[t] * inv;
    }
}

// out[b, i] = x3[b, nd_i] * prod_{j < min(nd_i,4)} x3[b, 5 + d_j(i) + 10j]
// Per-thread: index math once per i, reused across 128 batch rows from LDS.
__global__ void expand_kernel(const float* __restrict__ x3, float* __restrict__ out) {
    __shared__ float sx[128 * 56];            // 28 KB: padded stride 56, slot 55 = 1.0
    int tid = threadIdx.x;
    int b_base = blockIdx.y * 128;

    if (tid < 128) sx[tid * 56 + 55] = 1.0f;  // neutral factor for invalid j

#ifdef USE_TDM
    // TDM iterate mode: tile = one 55-dword row; 128 iterations advancing
    // global by 55 dwords and LDS by 56 dwords (padded stride for bank spread).
    if (tid < 32) {                           // wave 0 issues; EXEC ignored by TDM
        unsigned long long ga = (unsigned long long)(uintptr_t)(x3 + (size_t)b_base * 55);
        unsigned lds0 = (unsigned)(uintptr_t)(void*)sx;   // flat[31:0] = LDS offset
        u32x4 g0;
        g0[0] = 1u;                                        // count=1, user mode
        g0[1] = lds0;                                      // lds_addr (bytes)
        g0[2] = (unsigned)ga;                              // global_addr[31:0]
        g0[3] = (unsigned)((ga >> 32) & 0x01FFFFFFu) | 0x80000000u;  // ga[56:32] | type=2
        i32x8 g1;
        g1[0] = (int)((2u << 16) | (1u << 19));            // data_size=4B, iterate_enable
        g1[1] = (int)(7040u << 16);                        // tensor_dim0=7040 (low 16) << 16
        g1[2] = (int)(128u << 16);                         // tensor_dim0 hi=0 | tensor_dim1=128
        g1[3] = (int)(55u << 16);                          // tensor_dim1 hi=0 | tile_dim0=55
        g1[4] = (int)1u;                                   // tile_dim1=1, tile_dim2=0
        g1[5] = (int)55u;                                  // tensor_dim0_stride[31:0]
        g1[6] = (int)(55u << 16);                          // stride0 hi=0 | tensor_dim1_stride lo
        g1[7] = 0;                                         // tensor_dim1_stride hi
        i32x4 g2;
        g2[0] = 1;                                         // tensor_dim2 (unused)
        g2[1] = 56;                                        // lds_addr_increment (elems)
        g2[2] = 55;                                        // global_addr_increment[31:0]
        g2[3] = (int)(127u << 16);                         // ga_inc hi=0 | iterate_count=127
        i32x4 g3 = { 0, 0, 0, 0 };
#if __clang_major__ >= 23
        i32x8 gx = { 0, 0, 0, 0, 0, 0, 0, 0 };             // clang-23 extra group (zeroed)
        __builtin_amdgcn_tensor_load_to_lds(g0, g1, g2, g3, gx, 0);
#else
        __builtin_amdgcn_tensor_load_to_lds(g0, g1, g2, g3, 0);
#endif
        __builtin_amdgcn_s_wait_tensorcnt(0);
    }
#else
    for (int idx = tid; idx < 128 * 55; idx += 256) {
        int r = idx / 55, cc = idx - r * 55;
        sx[r * 56 + cc] = x3[(b_base + r) * 55 + cc];
    }
#endif
    __syncthreads();

    unsigned i = blockIdx.x * 256u + (unsigned)tid;
    if (i >= NOUT) return;

    // digits via constant-divisor divides (mul-hi), no generic division
    unsigned q1 = i / 10u, q2 = i / 100u, q3 = i / 1000u, q4 = i / 10000u, q5 = i / 100000u;
    unsigned d0 = i  - q1 * 10u, d1 = q1 - q2 * 10u, d2 = q2 - q3 * 10u;
    unsigned d3 = q3 - q4 * 10u, d4 = q4 - q5 * 10u, d5 = 0u;  // i < 1e5 -> q5 = 0
    int n = 1 + (i >= 10u) + (i >= 100u) + (i >= 1000u) + (i >= 10000u);

    int off0n = n;                            // lead factor index: x3[b, n]
    int offj[4];
#pragma unroll
    for (int j = 0; j < 4; ++j) {
        int e = n - j; e = (e < 0) ? 0 : e;   // expo = max(n - j, 0)
        unsigned d = d5;
        d = (e == 4) ? d4 : d;
        d = (e == 3) ? d3 : d;
        d = (e == 2) ? d2 : d;
        d = (e == 1) ? d1 : d;
        d = (e == 0) ? d0 : d;
        offj[j] = (j < n) ? (int)(5u + d + 10u * (unsigned)j) : 55;  // 55 -> 1.0 slot
    }

    // Output is write-once and larger than L2 (205 MB > 192 MB): stream with
    // non-temporal stores so the write stream doesn't churn the L2.
    float* po = out + (size_t)b_base * NOUT + i;
    const float* row = sx;
#pragma unroll 4
    for (int b = 0; b < 128; ++b) {
        float v = row[off0n] * row[offj[0]] * row[offj[1]] * row[offj[2]] * row[offj[3]];
        __builtin_nontemporal_store(v, po);
        po  += NOUT;
        row += 56;
    }
}

extern "C" void kernel_launch(void* const* d_in, const int* in_sizes, int n_in,
                              void* d_out, int out_size, void* d_ws, size_t ws_size,
                              hipStream_t stream) {
    const float* x  = (const float*)d_in[0];
    const float* w2 = (const float*)d_in[1];
    const float* b2 = (const float*)d_in[2];
    const float* w3 = (const float*)d_in[3];
    const float* b3 = (const float*)d_in[4];
    float* out = (float*)d_out;

    char* ws = (char*)d_ws;
    _Float16* h16 = (_Float16*)ws;                       // 512*256*2  = 262144 B
    float*    z   = (float*)(ws + 262144);               // 512*55*4   = 112640 B
    float*    x3  = (float*)(ws + 262144 + 112640);      // 512*55*4   = 112640 B

    layer1_wmma<<<64, 256, 0, stream>>>(x, w2, b2, h16);
    layer2_wmma<<<16, 256, 0, stream>>>(h16, w3, b3, z);
    softmax_kernel<<<2, 256, 0, stream>>>(z, x3);
    dim3 g((NOUT + 255) / 256, 4);
    expand_kernel<<<g, 256, 0, stream>>>(x3, out);
}
